// SteerableEncoder_80066780332741
// MI455X (gfx1250) — matrix operand order repaired and checked by reading, not compiled, not run on hardware
//
#include <hip/hip_runtime.h>
#include <hip/hip_bf16.h>

typedef __attribute__((ext_vector_type(16))) __bf16 v16bf;
typedef __attribute__((ext_vector_type(8)))  __bf16 v8bf;
typedef __attribute__((ext_vector_type(8)))  float  v8f;

#define NA    1024
#define NGRID 8192
#define NN    9216
#define NE    204800
#define KA_   8
#define KG_   24
#define HDIM  240
#define HPAD  256
#define KPAD  512
#define NT    15

__device__ __forceinline__ float silu_f(float x) { return x / (1.0f + __expf(-x)); }

// ---------------- weight transpose: W (K x N, f32 row-major) -> Wt (N x Kpad, bf16, zero pad)
__global__ void wtrans_kernel(const float* __restrict__ W, __bf16* __restrict__ Wt,
                              int K, int N, int Kpad) {
  int idx = blockIdx.x * 256 + threadIdx.x;
  if (idx >= N * Kpad) return;
  int n = idx / Kpad, k = idx % Kpad;
  float v = (k < K) ? W[(long)k * N + n] : 0.0f;
  Wt[idx] = (__bf16)v;
}

// ---------------- zero f32 buffer
__global__ void zerof_kernel(float* __restrict__ p, long n) {
  long i = (long)blockIdx.x * 256 + threadIdx.x;
  long stride = (long)gridDim.x * 256;
  for (; i < n; i += stride) p[i] = 0.0f;
}

// ---------------- node positions: atoms then 16^3 grid x B
__global__ void nodepos_kernel(const float* __restrict__ pos, float* __restrict__ np_) {
  int i = blockIdx.x * 256 + threadIdx.x;
  if (i >= NN) return;
  if (i < NA) {
    np_[3*i+0] = pos[3*i+0]; np_[3*i+1] = pos[3*i+1]; np_[3*i+2] = pos[3*i+2];
  } else {
    int g  = i - NA;
    int gg = g & 4095;              // within-batch grid index
    int ix = gg >> 8, iy = (gg >> 4) & 15, iz = gg & 15;
    np_[3*i+0] = ((float)ix - 7.5f) * 1.5f;
    np_[3*i+1] = ((float)iy - 7.5f) * 1.5f;
    np_[3*i+2] = ((float)iz - 7.5f) * 1.5f;
  }
}

// ---------------- atom-atom kNN (top KA by d2, stable ties)
__global__ void knn_atoms_kernel(const float* __restrict__ pos, const int* __restrict__ ab,
                                 int* __restrict__ nbr) {
  int i = blockIdx.x * 256 + threadIdx.x;
  if (i >= NA) return;
  float px = pos[3*i], py = pos[3*i+1], pz = pos[3*i+2];
  int bi = ab[i];
  float bd[KA_]; int bx[KA_];
#pragma unroll
  for (int k = 0; k < KA_; ++k) { bd[k] = 3.0e38f; bx[k] = 0; }
  for (int j = 0; j < NA; ++j) {
    float dx = px - pos[3*j], dy = py - pos[3*j+1], dz = pz - pos[3*j+2];
    float d2 = dx*dx + dy*dy + dz*dz;
    if (j == i || ab[j] != bi) d2 = 3.0e38f;
    if (d2 < bd[KA_-1]) {
      int k = KA_ - 1;
      while (k > 0 && bd[k-1] > d2) { bd[k] = bd[k-1]; bx[k] = bx[k-1]; --k; }
      bd[k] = d2; bx[k] = j;
    }
  }
#pragma unroll
  for (int k = 0; k < KA_; ++k) nbr[i*KA_ + k] = bx[k];
}

// ---------------- grid-atom kNN (top KG), atoms staged via LDS
__global__ void knn_grid_kernel(const float* __restrict__ np_, const float* __restrict__ pos,
                                const int* __restrict__ ab, int* __restrict__ nbg) {
  __shared__ float sx[256], sy[256], sz[256];
  __shared__ int   sb[256];
  int tid = threadIdx.x;
  int gi  = blockIdx.x * 256 + tid;              // grid point 0..8191 (exact)
  float px = np_[3*(NA+gi)], py = np_[3*(NA+gi)+1], pz = np_[3*(NA+gi)+2];
  int bi = gi >> 12;                              // batch = gi / 4096
  float bd[KG_]; int bx[KG_];
#pragma unroll
  for (int k = 0; k < KG_; ++k) { bd[k] = 3.0e38f; bx[k] = 0; }
  for (int ch = 0; ch < NA / 256; ++ch) {
    int j = ch * 256 + tid;
    sx[tid] = pos[3*j]; sy[tid] = pos[3*j+1]; sz[tid] = pos[3*j+2]; sb[tid] = ab[j];
    __syncthreads();
    for (int jj = 0; jj < 256; ++jj) {
      float dx = px - sx[jj], dy = py - sy[jj], dz = pz - sz[jj];
      float d2 = dx*dx + dy*dy + dz*dz;
      if (sb[jj] != bi) d2 = 3.0e38f;
      if (d2 < bd[KG_-1]) {
        int k = KG_ - 1;
        while (k > 0 && bd[k-1] > d2) { bd[k] = bd[k-1]; bx[k] = bx[k-1]; --k; }
        bd[k] = d2; bx[k] = ch * 256 + jj;
      }
    }
    __syncthreads();
  }
#pragma unroll
  for (int k = 0; k < KG_; ++k) nbg[gi*KG_ + k] = bx[k];
}

// ---------------- edge features: sph harmonics * mask -> bf16 (padded 32), segment sums
__global__ void edge_kernel(const float* __restrict__ np_,
                            const int* __restrict__ nbr, const int* __restrict__ nbg,
                            __bf16* __restrict__ eab, float* __restrict__ maskE,
                            float* __restrict__ na_sum, float* __restrict__ cnt) {
  long e = (long)blockIdx.x * 256 + threadIdx.x;
  if (e >= NE) return;
  int src, dst;
  if (e < (long)NA * KA_) { dst = (int)(e >> 3); src = nbr[e]; }
  else { long ge = e - (long)NA * KA_; dst = NA + (int)(ge / KG_); src = nbg[ge]; }
  float rx = np_[3*src]   - np_[3*dst];
  float ry = np_[3*src+1] - np_[3*dst+1];
  float rz = np_[3*src+2] - np_[3*dst+2];
  float dist = __builtin_sqrtf(rx*rx + ry*ry + rz*rz);
  float mask = (dist <= 8.0f) ? 1.0f : 0.0f;
  float inv  = 1.0f / fmaxf(dist, 1e-9f);
  float x = rx * inv, y = ry * inv, z = rz * inv;
  float s[9];
  s[0] = 0.28209479177387814f;
  s[1] = 0.4886025119029199f * y;
  s[2] = 0.4886025119029199f * z;
  s[3] = 0.4886025119029199f * x;
  s[4] = 1.0925484305920792f * x * y;
  s[5] = 1.0925484305920792f * y * z;
  s[6] = 0.31539156525252005f * (3.0f * z * z - 1.0f);
  s[7] = 1.0925484305920792f * z * x;
  s[8] = 0.5462742152960396f * (x * x - y * y);
#pragma unroll
  for (int c = 0; c < 9; ++c) {
    float v = s[c] * mask;
    eab[e*32 + c] = (__bf16)v;
    atomicAdd(&na_sum[(long)dst * 9 + c], v);
  }
#pragma unroll
  for (int c = 9; c < 32; ++c) eab[e*32 + c] = (__bf16)0.0f;
  maskE[e] = mask;
  atomicAdd(&cnt[dst], mask);
}

// ---------------- node_attr finalize + embedding h = silu([onehot|attr] @ We + be)
__global__ void embed_kernel(const int* __restrict__ atype,
                             const float* __restrict__ na_sum, const float* __restrict__ cnt,
                             const float* __restrict__ We, const float* __restrict__ be,
                             float* __restrict__ node_attr,
                             float* __restrict__ h, __bf16* __restrict__ hb) {
  __shared__ float na[9];
  int i = blockIdx.x;              // node
  int t = threadIdx.x;
  if (t < 9) {
    float v = (t == 0) ? 1.0f : na_sum[(long)i*9 + t] / fmaxf(cnt[i], 1.0f);
    na[t] = v;
    node_attr[(long)i*9 + t] = v;
  }
  __syncthreads();
  if (t < HDIM) {
    float acc = be[t];
    if (i < NA) acc += We[(long)atype[i] * HDIM + t];
#pragma unroll
    for (int c = 0; c < 9; ++c) acc += na[c] * We[(long)(16 + c) * HDIM + t];
    float v = silu_f(acc);
    h[(long)i * HDIM + t]  = v;
    hb[(long)i * HDIM + t] = (__bf16)v;
  }
}

// ---------------- fused message MLP: silu(silu([h_src|h_dst|ea] W1 + b1) W2 + b2)*mask -> atomic agg[dst]
__global__ __launch_bounds__(256)
void msg_kernel(const __bf16* __restrict__ hb, const __bf16* __restrict__ eab,
                const float* __restrict__ maskE,
                const int* __restrict__ nbr, const int* __restrict__ nbg,
                const __bf16* __restrict__ W1t, const float* __restrict__ b1,
                const __bf16* __restrict__ W2t, const float* __restrict__ b2,
                float* __restrict__ agg) {
  __shared__ __bf16 Abuf[16][KPAD];   // 16 KB
  __shared__ __bf16 Ybuf[16][HPAD];   // 8 KB
  __shared__ int   srcS[16], dstS[16];
  __shared__ float maskS[16];

  const int tid  = threadIdx.x;
  const int lane = tid & 31;
  const int wave = tid >> 5;
  const int col  = lane & 15;
  const int half = lane >> 4;
  const long e0  = (long)blockIdx.x * 16;

  if (tid < 16) {
    long e = e0 + tid;
    int s_, d_;
    if (e < (long)NA * KA_) { d_ = (int)(e >> 3); s_ = nbr[e]; }
    else { long ge = e - (long)NA * KA_; d_ = NA + (int)(ge / KG_); s_ = nbg[ge]; }
    srcS[tid] = s_; dstS[tid] = d_; maskS[tid] = maskE[e];
  }
  __syncthreads();

  // stage A panel: 16 rows x 512 cols, 8-bf16 (16B) chunks; 1024 tasks / 256 threads
#pragma unroll
  for (int q = tid; q < 1024; q += 256) {
    int r  = q >> 6;
    int cc = q & 63;
    v8bf val;
    if (cc < 30)       val = *(const v8bf*)(hb + (long)srcS[r] * HDIM + cc * 8);
    else if (cc < 60)  val = *(const v8bf*)(hb + (long)dstS[r] * HDIM + (cc - 30) * 8);
    else               val = *(const v8bf*)(eab + (e0 + r) * 32 + (cc - 60) * 8);
    *(v8bf*)(&Abuf[r][cc * 8]) = val;
  }
  Ybuf[tid >> 4][HDIM + (tid & 15)] = (__bf16)0.0f;   // zero K-pad cols of Y
  __syncthreads();

  // GEMM1: (16 x 512) x (512 x 240)
  for (int nt = wave; nt < NT; nt += 8) {
    int ncol = nt * 16 + col;
    v8f c = {0.f, 0.f, 0.f, 0.f, 0.f, 0.f, 0.f, 0.f};
#pragma unroll
    for (int s = 0; s < KPAD / 32; ++s) {
      v16bf a, b;
      const __bf16* ap = &Abuf[col][s * 32 + half * 8];
      ((v8bf*)&a)[0] = *(const v8bf*)ap;
      ((v8bf*)&a)[1] = *(const v8bf*)(ap + 16);
      ((v8bf*)&b)[0] = *(const v8bf*)(W1t + (long)ncol * KPAD + s * 32 + half * 8);
      ((v8bf*)&b)[1] = *(const v8bf*)(W1t + (long)ncol * KPAD + s * 32 + half * 8 + 16);
      c = __builtin_amdgcn_wmma_f32_16x16x32_bf16(false, a, false, b, (short)0, c, false, false);
    }
    float bias = b1[ncol];
#pragma unroll
    for (int v = 0; v < 8; ++v)
      Ybuf[v + half * 8][ncol] = (__bf16)silu_f(c[v] + bias);
  }
  __syncthreads();

  // GEMM2: (16 x 256) x (256 x 240), epilogue: silu * mask -> atomic scatter to agg[dst]
  for (int nt = wave; nt < NT; nt += 8) {
    int ncol = nt * 16 + col;
    v8f c = {0.f, 0.f, 0.f, 0.f, 0.f, 0.f, 0.f, 0.f};
#pragma unroll
    for (int s = 0; s < HPAD / 32; ++s) {
      v16bf a, b;
      const __bf16* ap = &Ybuf[col][s * 32 + half * 8];
      ((v8bf*)&a)[0] = *(const v8bf*)ap;
      ((v8bf*)&a)[1] = *(const v8bf*)(ap + 16);
      ((v8bf*)&b)[0] = *(const v8bf*)(W2t + (long)ncol * HPAD + s * 32 + half * 8);
      ((v8bf*)&b)[1] = *(const v8bf*)(W2t + (long)ncol * HPAD + s * 32 + half * 8 + 16);
      c = __builtin_amdgcn_wmma_f32_16x16x32_bf16(false, a, false, b, (short)0, c, false, false);
    }
    float bias = b2[ncol];
#pragma unroll
    for (int v = 0; v < 8; ++v) {
      int m = v + half * 8;
      float y = silu_f(c[v] + bias) * maskS[m];
      atomicAdd(agg + (long)dstS[m] * HDIM + ncol, y);
    }
  }
}

// ---------------- fused update MLP: h += silu(silu([h|agg|attr] W1 + b1) W2 + b2)
__global__ __launch_bounds__(256)
void upd_kernel(const float* __restrict__ node_attr, const float* __restrict__ agg,
                const __bf16* __restrict__ W1t, const float* __restrict__ b1,
                const __bf16* __restrict__ W2t, const float* __restrict__ b2,
                float* __restrict__ h, __bf16* __restrict__ hb) {
  __shared__ __bf16 Abuf[16][KPAD];
  __shared__ __bf16 Ybuf[16][HPAD];

  const int tid  = threadIdx.x;
  const int lane = tid & 31;
  const int wave = tid >> 5;
  const int col  = lane & 15;
  const int half = lane >> 4;
  const long i0  = (long)blockIdx.x * 16;

#pragma unroll
  for (int q = tid; q < 1024; q += 256) {
    int r  = q >> 6;
    int cc = q & 63;
    long i = i0 + r;
    __bf16* dp = &Abuf[r][cc * 8];
    if (cc < 30) {
      *(v8bf*)dp = *(const v8bf*)(hb + i * HDIM + cc * 8);
    } else if (cc < 60) {
      const float* ap = agg + i * HDIM + (cc - 30) * 8;
#pragma unroll
      for (int j = 0; j < 8; ++j) dp[j] = (__bf16)ap[j];
    } else if (cc == 60) {
#pragma unroll
      for (int j = 0; j < 8; ++j) dp[j] = (__bf16)node_attr[i * 9 + j];
    } else if (cc == 61) {
      dp[0] = (__bf16)node_attr[i * 9 + 8];
#pragma unroll
      for (int j = 1; j < 8; ++j) dp[j] = (__bf16)0.0f;
    } else {
#pragma unroll
      for (int j = 0; j < 8; ++j) dp[j] = (__bf16)0.0f;
    }
  }
  Ybuf[tid >> 4][HDIM + (tid & 15)] = (__bf16)0.0f;
  __syncthreads();

  for (int nt = wave; nt < NT; nt += 8) {
    int ncol = nt * 16 + col;
    v8f c = {0.f, 0.f, 0.f, 0.f, 0.f, 0.f, 0.f, 0.f};
#pragma unroll
    for (int s = 0; s < KPAD / 32; ++s) {
      v16bf a, b;
      const __bf16* ap = &Abuf[col][s * 32 + half * 8];
      ((v8bf*)&a)[0] = *(const v8bf*)ap;
      ((v8bf*)&a)[1] = *(const v8bf*)(ap + 16);
      ((v8bf*)&b)[0] = *(const v8bf*)(W1t + (long)ncol * KPAD + s * 32 + half * 8);
      ((v8bf*)&b)[1] = *(const v8bf*)(W1t + (long)ncol * KPAD + s * 32 + half * 8 + 16);
      c = __builtin_amdgcn_wmma_f32_16x16x32_bf16(false, a, false, b, (short)0, c, false, false);
    }
    float bias = b1[ncol];
#pragma unroll
    for (int v = 0; v < 8; ++v)
      Ybuf[v + half * 8][ncol] = (__bf16)silu_f(c[v] + bias);
  }
  __syncthreads();

  for (int nt = wave; nt < NT; nt += 8) {
    int ncol = nt * 16 + col;
    v8f c = {0.f, 0.f, 0.f, 0.f, 0.f, 0.f, 0.f, 0.f};
#pragma unroll
    for (int s = 0; s < HPAD / 32; ++s) {
      v16bf a, b;
      const __bf16* ap = &Ybuf[col][s * 32 + half * 8];
      ((v8bf*)&a)[0] = *(const v8bf*)ap;
      ((v8bf*)&a)[1] = *(const v8bf*)(ap + 16);
      ((v8bf*)&b)[0] = *(const v8bf*)(W2t + (long)ncol * HPAD + s * 32 + half * 8);
      ((v8bf*)&b)[1] = *(const v8bf*)(W2t + (long)ncol * HPAD + s * 32 + half * 8 + 16);
      c = __builtin_amdgcn_wmma_f32_16x16x32_bf16(false, a, false, b, (short)0, c, false, false);
    }
    float bias = b2[ncol];
#pragma unroll
    for (int v = 0; v < 8; ++v) {
      long i = i0 + v + half * 8;
      float hn = h[i * HDIM + ncol] + silu_f(c[v] + bias);
      h[i * HDIM + ncol]  = hn;
      hb[i * HDIM + ncol] = (__bf16)hn;
    }
  }
}

// ---------------- readout for grid nodes: out = silu(h Wo1 + bo1) Wo2 + bo2
__global__ __launch_bounds__(256)
void out_kernel(const __bf16* __restrict__ hb,
                const __bf16* __restrict__ W1t, const float* __restrict__ b1,
                const __bf16* __restrict__ W2t, const float* __restrict__ b2,
                float* __restrict__ out) {
  __shared__ __bf16 Abuf[16][HPAD];
  __shared__ __bf16 Ybuf[16][HPAD];

  const int tid  = threadIdx.x;
  const int lane = tid & 31;
  const int wave = tid >> 5;
  const int col  = lane & 15;
  const int half = lane >> 4;
  const long g0  = (long)blockIdx.x * 16;   // grid-node tile

#pragma unroll
  for (int q = tid; q < 512; q += 256) {     // 16 rows x 32 chunks of 8
    int r  = q >> 5;
    int cc = q & 31;
    __bf16* dp = &Abuf[r][cc * 8];
    if (cc < 30) {
      *(v8bf*)dp = *(const v8bf*)(hb + (NA + g0 + r) * (long)HDIM + cc * 8);
    } else {
#pragma unroll
      for (int j = 0; j < 8; ++j) dp[j] = (__bf16)0.0f;
    }
  }
  Ybuf[tid >> 4][HDIM + (tid & 15)] = (__bf16)0.0f;
  __syncthreads();

  for (int nt = wave; nt < NT; nt += 8) {
    int ncol = nt * 16 + col;
    v8f c = {0.f, 0.f, 0.f, 0.f, 0.f, 0.f, 0.f, 0.f};
#pragma unroll
    for (int s = 0; s < HPAD / 32; ++s) {
      v16bf a, b;
      const __bf16* ap = &Abuf[col][s * 32 + half * 8];
      ((v8bf*)&a)[0] = *(const v8bf*)ap;
      ((v8bf*)&a)[1] = *(const v8bf*)(ap + 16);
      ((v8bf*)&b)[0] = *(const v8bf*)(W1t + (long)ncol * HPAD + s * 32 + half * 8);
      ((v8bf*)&b)[1] = *(const v8bf*)(W1t + (long)ncol * HPAD + s * 32 + half * 8 + 16);
      c = __builtin_amdgcn_wmma_f32_16x16x32_bf16(false, a, false, b, (short)0, c, false, false);
    }
    float bias = b1[ncol];
#pragma unroll
    for (int v = 0; v < 8; ++v)
      Ybuf[v + half * 8][ncol] = (__bf16)silu_f(c[v] + bias);
  }
  __syncthreads();

  for (int nt = wave; nt < 2; nt += 8) {     // 32 output channels = 2 tiles
    int ncol = nt * 16 + col;
    v8f c = {0.f, 0.f, 0.f, 0.f, 0.f, 0.f, 0.f, 0.f};
#pragma unroll
    for (int s = 0; s < HPAD / 32; ++s) {
      v16bf a, b;
      const __bf16* ap = &Ybuf[col][s * 32 + half * 8];
      ((v8bf*)&a)[0] = *(const v8bf*)ap;
      ((v8bf*)&a)[1] = *(const v8bf*)(ap + 16);
      ((v8bf*)&b)[0] = *(const v8bf*)(W2t + (long)ncol * HPAD + s * 32 + half * 8);
      ((v8bf*)&b)[1] = *(const v8bf*)(W2t + (long)ncol * HPAD + s * 32 + half * 8 + 16);
      c = __builtin_amdgcn_wmma_f32_16x16x32_bf16(false, a, false, b, (short)0, c, false, false);
    }
    float bias = b2[ncol];
#pragma unroll
    for (int v = 0; v < 8; ++v) {
      long g = g0 + v + half * 8;
      out[g * 32 + ncol] = c[v] + bias;
    }
  }
}

extern "C" void kernel_launch(void* const* d_in, const int* in_sizes, int n_in,
                              void* d_out, int out_size, void* d_ws, size_t ws_size,
                              hipStream_t stream) {
  (void)in_sizes; (void)n_in; (void)out_size; (void)ws_size;
  const float* pos    = (const float*)d_in[0];
  const int*   atype  = (const int*)d_in[1];
  const int*   abatch = (const int*)d_in[2];
  const float* We     = (const float*)d_in[3];
  const float* be     = (const float*)d_in[4];
  const float* Wm1    = (const float*)d_in[5];
  const float* bm1    = (const float*)d_in[6];
  const float* Wm2    = (const float*)d_in[7];
  const float* bm2    = (const float*)d_in[8];
  const float* Wu1    = (const float*)d_in[9];
  const float* bu1    = (const float*)d_in[10];
  const float* Wu2    = (const float*)d_in[11];
  const float* bu2    = (const float*)d_in[12];
  const float* Wo1    = (const float*)d_in[13];
  const float* bo1    = (const float*)d_in[14];
  const float* Wo2    = (const float*)d_in[15];
  const float* bo2    = (const float*)d_in[16];
  float* out = (float*)d_out;

  char* ws = (char*)d_ws;
  size_t off = 0;
  auto alloc = [&](size_t bytes) { size_t o = off; off += (bytes + 255) & ~(size_t)255; return o; };

  float* node_pos  = (float*)(ws + alloc((size_t)NN * 3 * 4));
  int*   nbr       = (int*)  (ws + alloc((size_t)NA * KA_ * 4));
  int*   nbg       = (int*)  (ws + alloc((size_t)NGRID * KG_ * 4));
  float* maskE     = (float*)(ws + alloc((size_t)NE * 4));
  float* na_sum    = (float*)(ws + alloc((size_t)(NN * 9 + NN) * 4));   // na_sum + cnt contiguous
  float* cnt       = na_sum + (size_t)NN * 9;
  float* node_attr = (float*)(ws + alloc((size_t)NN * 9 * 4));
  float* h         = (float*)(ws + alloc((size_t)NN * HDIM * 4));
  float* agg       = (float*)(ws + alloc((size_t)NN * HDIM * 4));
  __bf16* eab      = (__bf16*)(ws + alloc((size_t)NE * 32 * 2));
  __bf16* hb       = (__bf16*)(ws + alloc((size_t)NN * HDIM * 2));
  __bf16* wm1t     = (__bf16*)(ws + alloc((size_t)2 * HDIM * KPAD * 2));
  __bf16* wm2t     = (__bf16*)(ws + alloc((size_t)2 * HDIM * HPAD * 2));
  __bf16* wu1t     = (__bf16*)(ws + alloc((size_t)2 * HDIM * KPAD * 2));
  __bf16* wu2t     = (__bf16*)(ws + alloc((size_t)2 * HDIM * HPAD * 2));
  __bf16* wo1t     = (__bf16*)(ws + alloc((size_t)HDIM * HPAD * 2));
  __bf16* wo2t     = (__bf16*)(ws + alloc((size_t)32 * HPAD * 2));

  // --- weight prep (bf16, transposed, K-padded)
  for (int l = 0; l < 2; ++l) {
    wtrans_kernel<<<(HDIM * KPAD + 255) / 256, 256, 0, stream>>>(
        Wm1 + (size_t)l * 489 * HDIM, wm1t + (size_t)l * HDIM * KPAD, 489, HDIM, KPAD);
    wtrans_kernel<<<(HDIM * HPAD + 255) / 256, 256, 0, stream>>>(
        Wm2 + (size_t)l * HDIM * HDIM, wm2t + (size_t)l * HDIM * HPAD, HDIM, HDIM, HPAD);
    wtrans_kernel<<<(HDIM * KPAD + 255) / 256, 256, 0, stream>>>(
        Wu1 + (size_t)l * 489 * HDIM, wu1t + (size_t)l * HDIM * KPAD, 489, HDIM, KPAD);
    wtrans_kernel<<<(HDIM * HPAD + 255) / 256, 256, 0, stream>>>(
        Wu2 + (size_t)l * HDIM * HDIM, wu2t + (size_t)l * HDIM * HPAD, HDIM, HDIM, HPAD);
  }
  wtrans_kernel<<<(HDIM * HPAD + 255) / 256, 256, 0, stream>>>(Wo1, wo1t, HDIM, HDIM, HPAD);
  wtrans_kernel<<<(32 * HPAD + 255) / 256, 256, 0, stream>>>(Wo2, wo2t, HDIM, 32, HPAD);

  // --- geometry + neighbor lists
  nodepos_kernel<<<(NN + 255) / 256, 256, 0, stream>>>(pos, node_pos);
  zerof_kernel<<<256, 256, 0, stream>>>(na_sum, (long)(NN * 9 + NN));
  knn_atoms_kernel<<<NA / 256, 256, 0, stream>>>(pos, abatch, nbr);
  knn_grid_kernel<<<NGRID / 256, 256, 0, stream>>>(node_pos, pos, abatch, nbg);

  // --- edge features + node_attr accumulation
  edge_kernel<<<NE / 256, 256, 0, stream>>>(node_pos, nbr, nbg, eab, maskE, na_sum, cnt);

  // --- embedding
  embed_kernel<<<NN, 256, 0, stream>>>(atype, na_sum, cnt, We, be, node_attr, h, hb);

  // --- message passing layers
  for (int l = 0; l < 2; ++l) {
    zerof_kernel<<<512, 256, 0, stream>>>(agg, (long)NN * HDIM);
    msg_kernel<<<NE / 16, 256, 0, stream>>>(hb, eab, maskE, nbr, nbg,
        wm1t + (size_t)l * HDIM * KPAD, bm1 + (size_t)l * HDIM,
        wm2t + (size_t)l * HDIM * HPAD, bm2 + (size_t)l * HDIM, agg);
    upd_kernel<<<NN / 16, 256, 0, stream>>>(node_attr, agg,
        wu1t + (size_t)l * HDIM * KPAD, bu1 + (size_t)l * HDIM,
        wu2t + (size_t)l * HDIM * HPAD, bu2 + (size_t)l * HDIM, h, hb);
  }

  // --- readout (grid nodes only)
  out_kernel<<<NGRID / 16, 256, 0, stream>>>(hb, wo1t, bo1, wo2t, bo2, out);
}
